// ImprovedEEGNet_54004918780587
// MI455X (gfx1250) — matrix operand order repaired and checked by reading, not compiled, not run on hardware
//
#include <hip/hip_runtime.h>
#include <hip/hip_bf16.h>

// ImprovedEEGNet for MI455X (gfx1250, wave32, WMMA).
// All conv / FC math routed through v_wmma_f32_16x16x32_f16 (f16 in, f32 acc).
// conv1 = implicit GEMM staged in LDS via the Tensor Data Mover (TDM), other
// convs use K-major im2col f16 buffers so each WMMA B-fragment load is one
// contiguous 32B read per lane. Generic GEMM uses 64x64-per-wave tiles
// (16 WMMA / K-step, ~32 FLOP per fragment byte). Subject-BN done exactly as
// reference: masked sums / full denominator, shift summed over all subjects.

typedef __attribute__((ext_vector_type(16))) _Float16 v16h;
typedef __attribute__((ext_vector_type(8)))  float    v8f;
typedef unsigned int v4u __attribute__((ext_vector_type(4)));
typedef int v8i __attribute__((ext_vector_type(8)));
typedef int v4i __attribute__((ext_vector_type(4)));

#define NSUBJ 4
#define CH 271
#define BN_EPS 1e-5f

__device__ __forceinline__ float eluf(float v) { return v > 0.f ? v : (expf(v) - 1.f); }

__device__ __forceinline__ v8f wmma32(v16h a, v16h b, v8f c) {
  return __builtin_amdgcn_wmma_f32_16x16x32_f16(false, a, false, b, (short)0, c, false, false);
}

// A fragment: row-major f16 [M][lda]; lanes0-15 row=m0+lane,
// halves h = K k0+h (lanes0-15) or k0+16+h (lanes16-31).
__device__ __forceinline__ v16h load_a_frag(const _Float16* A, int lda, int m0, int k0) {
  const int lane = threadIdx.x & 31;
  const _Float16* p = A + (size_t)(m0 + (lane & 15)) * lda + k0 + ((lane & 16) ? 16 : 0);
  v16h f;
#pragma unroll
  for (int i = 0; i < 16; ++i) f[i] = p[i];
  return f;
}

// B fragment from K-major ("transposed") buffer Bt[N][ldb]: lane holds column
// n0+(lane&15), halves h = K k0+h (lanes0-15) or k0+16+h (lanes16-31): one
// contiguous 32-byte read per lane.
__device__ __forceinline__ v16h load_bt_frag(const _Float16* Bt, int ldb, int n0, int k0) {
  const int lane = threadIdx.x & 31;
  const _Float16* p = Bt + (size_t)(n0 + (lane & 15)) * ldb + k0 + ((lane & 16) ? 16 : 0);
  v16h f;
#pragma unroll
  for (int i = 0; i < 16; ++i) f[i] = p[i];
  return f;
}

// C layout (16x16 f32): VGPR r, lanes0-15: M=r, N=lane; lanes16-31: M=8+r.
__device__ __forceinline__ void store_c_f32(float* C, int ldc, int m0, int n0, v8f c) {
  const int lane = threadIdx.x & 31;
  float* p = C + (size_t)(m0 + ((lane & 16) ? 8 : 0)) * ldc + n0 + (lane & 15);
#pragma unroll
  for (int r = 0; r < 8; ++r) p[(size_t)r * ldc] = c[r];
}

__device__ __forceinline__ void store_c_f16(_Float16* C, size_t ldc, int m0, int n0, v8f c) {
  const int lane = threadIdx.x & 31;
  _Float16* p = C + (size_t)(m0 + ((lane & 16) ? 8 : 0)) * ldc + n0 + (lane & 15);
#pragma unroll
  for (int r = 0; r < 8; ++r) p[(size_t)r * ldc] = (_Float16)c[r];
}

// ---------------- Generic WMMA GEMM: C[M][N] = A[M][K] * Bt[N][K]^T --------
// One wave per 64x64 C tile (16 accumulators); M,N mult of 64, K mult of 32.
__global__ __launch_bounds__(256) void k_gemm_wmma(
    const _Float16* __restrict__ A, const _Float16* __restrict__ Bt,
    float* __restrict__ C, int M, int N, int K) {
  int wv = (blockIdx.x * blockDim.x + threadIdx.x) >> 5;
  int tm = M >> 6, tn = N >> 6;
  if (wv >= tm * tn) return;                 // wave-uniform: EXEC stays full
  int m0 = (wv % tm) * 64, n0 = (wv / tm) * 64;
  v8f acc[16];
#pragma unroll
  for (int i = 0; i < 16; ++i) acc[i] = {};
#pragma unroll 1
  for (int k0 = 0; k0 < K; k0 += 32) {
    v16h af[4], bf[4];
#pragma unroll
    for (int i = 0; i < 4; ++i) af[i] = load_a_frag(A, K, m0 + 16 * i, k0);
#pragma unroll
    for (int j = 0; j < 4; ++j) bf[j] = load_bt_frag(Bt, K, n0 + 16 * j, k0);
#pragma unroll
    for (int i = 0; i < 4; ++i)
#pragma unroll
      for (int j = 0; j < 4; ++j) acc[i * 4 + j] = wmma32(af[i], bf[j], acc[i * 4 + j]);
  }
#pragma unroll
  for (int i = 0; i < 4; ++i)
#pragma unroll
    for (int j = 0; j < 4; ++j) store_c_f32(C, N, m0 + 16 * i, n0 + 16 * j, acc[i * 4 + j]);
}

// ---------------- conv1: implicit GEMM in LDS, row staged by TDM -----------
// One block per (b,h) row. Wave 0 issues a TENSOR_LOAD_TO_LDS of the 256-f32
// row (1-D D#), overlaps the weight-bank fill, then s_wait_tensorcnt before
// the barrier. A = weights 32x51 zero-padded to 32x64 (LDS); implicit B = row
// with halo. 8 waves tile the 32x256 output with WMMA.
__global__ __launch_bounds__(256) void k_conv1_wmma(
    const float* __restrict__ x, const float* __restrict__ w51,
    _Float16* __restrict__ h1) {
  __shared__ float xsf[256];         // TDM target (raw f32 row)
  __shared__ _Float16 xs[352];       // w = -32 .. 319 (zero outside [0,256))
  __shared__ _Float16 wl[32 * 64];   // [oc][k], zero-padded k=51..63
  int row = blockIdx.x;
  int b = row / CH, hh = row % CH;
  const float* xp = x + ((size_t)b * CH + hh) * 256;
  int t = threadIdx.x;

  if (t < 32) {
    // D# group0: count=1 | lds_addr | global_addr[56:0] | type=2
    unsigned long long ga = (unsigned long long)(uintptr_t)xp;
    unsigned int la = (unsigned int)(uintptr_t)xsf;  // low 32b of generic = LDS offset
    v4u g0 = { 1u, la, (unsigned int)ga,
               (unsigned int)((ga >> 32) & 0x01FFFFFFu) | 0x80000000u };
    // D# group1: data_size=4B | tensor_dim0=256 | tensor_dim1=1 |
    //            tile_dim0=256 | tile_dim1=1 | dim0_stride=256
    v8i g1 = { (int)(2u << 16), (int)(256u << 16), (int)(1u << 16),
               (int)(256u << 16), 1, 256, 0, 0 };
    v4i z4 = { 0, 0, 0, 0 };
    v8i z8 = { 0, 0, 0, 0, 0, 0, 0, 0 };
    __builtin_amdgcn_tensor_load_to_lds(g0, g1, z4, z4, z8, 0);
  }
  for (int i = t; i < 32 * 64; i += 256) {        // overlap with TDM
    int oc = i >> 6, k = i & 63;
    wl[i] = (k < 51) ? (_Float16)w51[oc * 51 + k] : (_Float16)0.f;
  }
  if (t < 32) __builtin_amdgcn_s_wait_tensorcnt(0);
  __syncthreads();
  for (int i = t; i < 352; i += 256) {            // f32 -> f16 with halo
    int wp = i - 32;
    xs[i] = (wp >= 0 && wp < 256) ? (_Float16)xsf[wp] : (_Float16)0.f;
  }
  __syncthreads();

  int wave = t >> 5, lane = t & 31;
  int n0 = wave * 32;                 // this wave's output w range [n0, n0+32)
  v8f c00 = {}, c01 = {}, c10 = {}, c11 = {};
  for (int ks = 0; ks < 64; ks += 32) {
    v16h a0 = load_a_frag(wl, 64, 0, ks);
    v16h a1 = load_a_frag(wl, 64, 16, ks);
    // implicit im2col B: B[k][n] = xs[32 + n + k - 25]; per-lane contiguous
    const _Float16* p0 = xs + 7 + n0 + (lane & 15) + ks + ((lane & 16) ? 16 : 0);
    v16h b0, b1;
#pragma unroll
    for (int i = 0; i < 16; ++i) { b0[i] = p0[i]; b1[i] = p0[16 + i]; }
    c00 = wmma32(a0, b0, c00);
    c01 = wmma32(a0, b1, c01);
    c10 = wmma32(a1, b0, c10);
    c11 = wmma32(a1, b1, c11);
  }
  _Float16* base = h1 + ((size_t)b * 32 * CH + hh) * 256;
  size_t ldc = (size_t)CH * 256;
  store_c_f16(base, ldc, 0, n0, c00);
  store_c_f16(base, ldc, 0, n0 + 16, c01);
  store_c_f16(base, ldc, 16, n0, c10);
  store_c_f16(base, ldc, 16, n0 + 16, c11);
}

// ---------------- subject-BN statistics ------------------------------------
// f16 [B][C][HW] source (conv1 output); one block per (b,c).
__global__ __launch_bounds__(256) void k_stats_bchw_f16(
    const _Float16* __restrict__ h, int C, int HW, const int* __restrict__ subj,
    float* __restrict__ sum, float* __restrict__ sq) {
  __shared__ float ssum, ssq;
  int b = blockIdx.x / C, c = blockIdx.x % C;
  if (threadIdx.x == 0) { ssum = 0.f; ssq = 0.f; }
  __syncthreads();
  const _Float16* p = h + ((size_t)b * C + c) * HW;
  float s = 0.f, q = 0.f;
  for (int i = threadIdx.x; i < HW; i += 256) {
    float v = (float)p[i];
    s += v; q += v * v;
  }
  atomicAdd(&ssum, s);
  atomicAdd(&ssq, q);
  __syncthreads();
  if (threadIdx.x == 0) {
    int su = subj[b];
    atomicAdd(&sum[su * C + c], ssum);
    atomicAdd(&sq[su * C + c], ssq);
  }
}

// f32 [OC][N] source (GEMM output, n = b*W + w); one block per oc.
__global__ __launch_bounds__(256) void k_stats_cn_f32(
    const float* __restrict__ Cm, int OC, int N, int W,
    const int* __restrict__ subj, float* __restrict__ sum, float* __restrict__ sq) {
  __shared__ float ss[NSUBJ], qq[NSUBJ];
  int oc = blockIdx.x;
  if (threadIdx.x < NSUBJ) { ss[threadIdx.x] = 0.f; qq[threadIdx.x] = 0.f; }
  __syncthreads();
  const float* p = Cm + (size_t)oc * N;
  float s[NSUBJ] = {0.f, 0.f, 0.f, 0.f}, q[NSUBJ] = {0.f, 0.f, 0.f, 0.f};
  for (int n = threadIdx.x; n < N; n += 256) {
    int su = subj[n / W];
    float v = p[n];
    s[su] += v; q[su] += v * v;
  }
#pragma unroll
  for (int i = 0; i < NSUBJ; ++i) { atomicAdd(&ss[i], s[i]); atomicAdd(&qq[i], q[i]); }
  __syncthreads();
  if (threadIdx.x < NSUBJ) {
    atomicAdd(&sum[threadIdx.x * OC + oc], ss[threadIdx.x]);
    atomicAdd(&sq[threadIdx.x * OC + oc], qq[threadIdx.x]);
  }
}

// inv[s][c] = gamma*rsqrt(var+eps), shift[c] = sum_s (beta - mean*inv)
__global__ void k_bn_coeffs(const float* __restrict__ sum, const float* __restrict__ sq,
                            const float* __restrict__ g, const float* __restrict__ bt,
                            float* __restrict__ inv, float* __restrict__ shift,
                            int C, float denom) {
  int c = blockIdx.x * blockDim.x + threadIdx.x;
  if (c >= C) return;
  float sh = 0.f;
  for (int s = 0; s < NSUBJ; ++s) {
    float mean = sum[s * C + c] / denom;
    float var = sq[s * C + c] / denom - mean * mean;
    float iv = g[s * C + c] * rsqrtf(var + BN_EPS);
    inv[s * C + c] = iv;
    sh += bt[s * C + c] - mean * iv;
  }
  shift[c] = sh;
}

// ---------------- BN apply + activation + pool -----------------------------
// bn1: elu + maxpool_w2 on f16 [B][32][271][256] -> f16 [B][32][271][128]
__global__ __launch_bounds__(256) void k_bn1_apply_maxpool(
    const _Float16* __restrict__ h1, const float* __restrict__ inv,
    const float* __restrict__ shift, const int* __restrict__ subj,
    _Float16* __restrict__ h1p) {
  int i = blockIdx.x * 256 + threadIdx.x;       // 128*32*271*128
  int w2 = i & 127;
  int r = i >> 7;
  int hh = r % CH; r /= CH;
  int c = r & 31;
  int b = r >> 5;
  if (b >= 128) return;
  const _Float16* p = h1 + ((size_t)(b * 32 + c) * CH + hh) * 256 + 2 * w2;
  float iv = inv[subj[b] * 32 + c], sh = shift[c];
  float a = eluf((float)p[0] * iv + sh);
  float d = eluf((float)p[1] * iv + sh);
  h1p[i] = (_Float16)fmaxf(a, d);
}

// depthwise (groups=32, kernel spans full H=271): out f32 [64][B*128]
__global__ __launch_bounds__(256) void k_dwconv(
    const _Float16* __restrict__ h1p, const float* __restrict__ dww,
    float* __restrict__ outp) {
  int i = blockIdx.x * 256 + threadIdx.x;       // 64 * 16384
  if (i >= 64 * 16384) return;
  int oc = i >> 14;
  int n = i & 16383;
  int b = n >> 7, w = n & 127;
  int ic = oc >> 1;
  const _Float16* p = h1p + (size_t)(b * 32 + ic) * CH * 128 + w;
  float acc = 0.f;
  for (int h = 0; h < CH; ++h) acc += dww[oc * CH + h] * (float)p[(size_t)h * 128];
  outp[i] = acc;
}

// generic: elu + avgpool_w2 on f32 [OC][128*W] -> f32 [128][OC][W/2]
__global__ __launch_bounds__(256) void k_bn_apply_avgpool(
    const float* __restrict__ Cm, int OC, int W, const float* __restrict__ inv,
    const float* __restrict__ shift, const int* __restrict__ subj,
    float* __restrict__ outp) {
  int W2 = W >> 1;
  int total = 128 * OC * W2;
  int i = blockIdx.x * 256 + threadIdx.x;
  if (i >= total) return;
  int w2 = i % W2;
  int r = i / W2;
  int oc = r % OC;
  int b = r / OC;
  int n = b * W + 2 * w2;
  float iv = inv[subj[b] * OC + oc], sh = shift[oc];
  const float* p = Cm + (size_t)oc * (128 * W) + n;
  float a = eluf(p[0] * iv + sh);
  float d = eluf(p[1] * iv + sh);
  outp[i] = 0.5f * (a + d);
}

// residual: h3 = elu(bn3(conv_res)) + h2, f32 [128][64][64]
__global__ __launch_bounds__(256) void k_res_apply(
    const float* __restrict__ Cm, const float* __restrict__ h2,
    const float* __restrict__ inv, const float* __restrict__ shift,
    const int* __restrict__ subj, float* __restrict__ h3) {
  int i = blockIdx.x * 256 + threadIdx.x;       // 128*64*64
  if (i >= 128 * 64 * 64) return;
  int w = i & 63, c = (i >> 6) & 63, b = i >> 12;
  int n = b * 64 + w;
  float iv = inv[subj[b] * 64 + c];
  h3[i] = eluf(Cm[(size_t)c * 8192 + n] * iv + shift[c]) + h2[i];
}

// attention gate: hatt = h6 * sigmoid(Cm)
__global__ __launch_bounds__(256) void k_attn_apply(
    const float* __restrict__ Cm, const float* __restrict__ h6,
    float* __restrict__ hatt) {
  int i = blockIdx.x * 256 + threadIdx.x;       // 128*512*8
  if (i >= 128 * 512 * 8) return;
  int w = i & 7, c = (i >> 3) & 511, b = i >> 12;
  float g = Cm[(size_t)c * 1024 + (b * 8 + w)];
  hatt[i] = h6[i] * (1.f / (1.f + expf(-g)));
}

// ---------------- packing kernels ------------------------------------------
// weights f32 [M][K] -> f16 [Mpad][Kpad] (zero fill)
__global__ __launch_bounds__(256) void k_pack_a(
    const float* __restrict__ src, _Float16* __restrict__ dst,
    int M, int K, int Mpad, int Kpad) {
  int i = blockIdx.x * 256 + threadIdx.x;
  if (i >= Mpad * Kpad) return;
  int m = i / Kpad, k = i % Kpad;
  dst[i] = (m < M && k < K) ? (_Float16)src[(size_t)m * K + k] : (_Float16)0.f;
}

// f32 [128][Cc][W] -> K-major im2col f16 Bt[(b*W+w)][c*KW+t]
__global__ __launch_bounds__(256) void k_pack_im2col(
    const float* __restrict__ h, _Float16* __restrict__ Bt,
    int Cc, int W, int KW, int P) {
  int K = Cc * KW;
  int total = 128 * W * K;
  int i = blockIdx.x * 256 + threadIdx.x;
  if (i >= total) return;
  int k = i % K, n = i / K;
  int b = n / W, w = n % W;
  int c = k / KW, t = k % KW;
  int wsrc = w + t - P;
  Bt[i] = (wsrc >= 0 && wsrc < W) ? (_Float16)h[((size_t)b * Cc + c) * W + wsrc]
                                  : (_Float16)0.f;
}

// z0t[b][k]: k<4096 -> flat hatt (c*8+w), 4096..4111 -> subject embedding, pad 0
__global__ __launch_bounds__(256) void k_build_z0(
    const float* __restrict__ hatt, const float* __restrict__ emb,
    const int* __restrict__ subj, _Float16* __restrict__ z0t) {
  int i = blockIdx.x * 256 + threadIdx.x;       // 128*4128
  if (i >= 128 * 4128) return;
  int b = i / 4128, k = i % 4128;
  float v = 0.f;
  if (k < 4096) v = hatt[(size_t)b * 4096 + k];
  else if (k < 4112) v = emb[subj[b] * 16 + (k - 4096)];
  z0t[i] = (_Float16)v;
}

// FC epilogue: elu(C^T + bias) -> f16 [128][M] (becomes next layer's Bt)
__global__ __launch_bounds__(256) void k_bias_elu_t(
    const float* __restrict__ Cm, const float* __restrict__ bias,
    _Float16* __restrict__ outT, int M) {
  int i = blockIdx.x * 256 + threadIdx.x;       // 128*M
  if (i >= 128 * M) return;
  int b = i / M, m = i % M;
  outT[i] = (_Float16)eluf(Cm[(size_t)m * 128 + b] + bias[m]);
}

__global__ __launch_bounds__(256) void k_final_out(
    const float* __restrict__ Cm, const float* __restrict__ bias,
    float* __restrict__ outp) {
  int i = blockIdx.x * 256 + threadIdx.x;       // 128*1854
  if (i >= 128 * 1854) return;
  int b = i / 1854, j = i % 1854;
  outp[i] = Cm[(size_t)j * 128 + b] + bias[j];
}

// ===========================================================================
extern "C" void kernel_launch(void* const* d_in, const int* in_sizes, int n_in,
                              void* d_out, int out_size, void* d_ws, size_t ws_size,
                              hipStream_t stream) {
  (void)in_sizes; (void)n_in; (void)out_size; (void)ws_size;
  const float* x       = (const float*)d_in[0];
  const float* conv1_w = (const float*)d_in[1];
  const float* bn1_g = (const float*)d_in[2];
  const float* bn1_b = (const float*)d_in[3];
  const float* dw_w  = (const float*)d_in[4];
  const float* bn2_g = (const float*)d_in[5];
  const float* bn2_b = (const float*)d_in[6];
  const float* res_w = (const float*)d_in[7];
  const float* bn3_g = (const float*)d_in[8];
  const float* bn3_b = (const float*)d_in[9];
  const float* sep1_w = (const float*)d_in[10];
  const float* bn4_g = (const float*)d_in[11];
  const float* bn4_b = (const float*)d_in[12];
  const float* sep2_w = (const float*)d_in[13];
  const float* bn5_g = (const float*)d_in[14];
  const float* bn5_b = (const float*)d_in[15];
  const float* sep3_w = (const float*)d_in[16];
  const float* bn6_g = (const float*)d_in[17];
  const float* bn6_b = (const float*)d_in[18];
  const float* attn_w = (const float*)d_in[19];
  const float* embed  = (const float*)d_in[20];
  const float* fc1_w = (const float*)d_in[21];
  const float* fc1_b = (const float*)d_in[22];
  const float* fc2_w = (const float*)d_in[23];
  const float* fc2_b = (const float*)d_in[24];
  const float* fc3_w = (const float*)d_in[25];
  const float* fc3_b = (const float*)d_in[26];
  const int* subj = (const int*)d_in[27];
  float* out = (float*)d_out;

  // Workspace carve-up (~900 MB peak; dominated by the f16 conv1 tensors).
  char* base = (char*)d_ws;
  size_t off = 0;
  auto take = [&](size_t bytes) -> char* {
    char* p = base + off;
    off = (off + bytes + 255) & ~(size_t)255;
    return p;
  };
  _Float16* h1   = (_Float16*)take((size_t)284164096 * 2);  // [128][32][271][256]
  _Float16* h1p  = (_Float16*)take((size_t)142082048 * 2);  // [128][32][271][128]
  float* ssum    = (float*)take(4 * 512 * 4);
  float* ssq     = (float*)take(4 * 512 * 4);
  float* binv    = (float*)take(4 * 512 * 4);
  float* bshift  = (float*)take(512 * 4);
  float* dwo     = (float*)take((size_t)64 * 16384 * 4);    // [64][B*128]
  float* h2      = (float*)take((size_t)524288 * 4);        // [128][64][64]
  float* h3      = (float*)take((size_t)524288 * 4);
  float* h4      = (float*)take((size_t)524288 * 4);        // [128][128][32]
  float* h5      = (float*)take((size_t)524288 * 4);        // [128][256][16]
  float* h6      = (float*)take((size_t)524288 * 4);        // [128][512][8]
  float* hatt    = (float*)take((size_t)524288 * 4);
  _Float16* Abuf  = (_Float16*)take((size_t)1024 * 4128 * 2);
  _Float16* Btbuf = (_Float16*)take((size_t)8192 * 960 * 2);
  float* Cbuf     = (float*)take((size_t)1048576 * 4);

  auto gemm = [&](const _Float16* A, const _Float16* Bt, float* C, int M, int N, int K) {
    int waves = (M / 64) * (N / 64);
    k_gemm_wmma<<<(waves + 7) / 8, 256, 0, stream>>>(A, Bt, C, M, N, K);
  };
  auto zstats = [&]() {
    (void)hipMemsetAsync(ssum, 0, 4 * 512 * 4, stream);
    (void)hipMemsetAsync(ssq, 0, 4 * 512 * 4, stream);
  };
  auto coeffs = [&](const float* g, const float* bb, int C, float denom) {
    k_bn_coeffs<<<(C + 63) / 64, 64, 0, stream>>>(ssum, ssq, g, bb, binv, bshift, C, denom);
  };

  // ---- conv1 (implicit GEMM, TDM + WMMA) + bn1 + elu + maxpool
  k_conv1_wmma<<<128 * CH, 256, 0, stream>>>(x, conv1_w, h1);
  zstats();
  k_stats_bchw_f16<<<128 * 32, 256, 0, stream>>>(h1, 32, CH * 256, subj, ssum, ssq);
  coeffs(bn1_g, bn1_b, 32, 128.f * 271.f * 256.f);
  k_bn1_apply_maxpool<<<142082048 / 256, 256, 0, stream>>>(h1, binv, bshift, subj, h1p);

  // ---- depthwise conv + bn2 + elu + avgpool
  k_dwconv<<<(64 * 16384) / 256, 256, 0, stream>>>(h1p, dw_w, dwo);
  zstats();
  k_stats_cn_f32<<<64, 256, 0, stream>>>(dwo, 64, 16384, 128, subj, ssum, ssq);
  coeffs(bn2_g, bn2_b, 64, 16384.f);
  k_bn_apply_avgpool<<<(128 * 64 * 64) / 256, 256, 0, stream>>>(dwo, 64, 128, binv, bshift, subj, h2);

  // ---- residual 1x1 conv + bn3 + elu + add
  k_pack_a<<<(64 * 64) / 256, 256, 0, stream>>>(res_w, Abuf, 64, 64, 64, 64);
  k_pack_im2col<<<(8192 * 64) / 256, 256, 0, stream>>>(h2, Btbuf, 64, 64, 1, 0);
  gemm(Abuf, Btbuf, Cbuf, 64, 8192, 64);
  zstats();
  k_stats_cn_f32<<<64, 256, 0, stream>>>(Cbuf, 64, 8192, 64, subj, ssum, ssq);
  coeffs(bn3_g, bn3_b, 64, 8192.f);
  k_res_apply<<<(128 * 64 * 64) / 256, 256, 0, stream>>>(Cbuf, h2, binv, bshift, subj, h3);

  // ---- sep1 (64->128, KW=15) + bn4 + elu + avgpool
  k_pack_a<<<(128 * 960) / 256, 256, 0, stream>>>(sep1_w, Abuf, 128, 960, 128, 960);
  k_pack_im2col<<<(8192 * 960) / 256, 256, 0, stream>>>(h3, Btbuf, 64, 64, 15, 7);
  gemm(Abuf, Btbuf, Cbuf, 128, 8192, 960);
  zstats();
  k_stats_cn_f32<<<128, 256, 0, stream>>>(Cbuf, 128, 8192, 64, subj, ssum, ssq);
  coeffs(bn4_g, bn4_b, 128, 8192.f);
  k_bn_apply_avgpool<<<(128 * 128 * 32) / 256, 256, 0, stream>>>(Cbuf, 128, 64, binv, bshift, subj, h4);

  // ---- sep2 (128->256, KW=15) + bn5 + elu + avgpool
  k_pack_a<<<(256 * 1920) / 256, 256, 0, stream>>>(sep2_w, Abuf, 256, 1920, 256, 1920);
  k_pack_im2col<<<(4096 * 1920) / 256, 256, 0, stream>>>(h4, Btbuf, 128, 32, 15, 7);
  gemm(Abuf, Btbuf, Cbuf, 256, 4096, 1920);
  zstats();
  k_stats_cn_f32<<<256, 256, 0, stream>>>(Cbuf, 256, 4096, 32, subj, ssum, ssq);
  coeffs(bn5_g, bn5_b, 256, 4096.f);
  k_bn_apply_avgpool<<<(128 * 256 * 16) / 256, 256, 0, stream>>>(Cbuf, 256, 32, binv, bshift, subj, h5);

  // ---- sep3 (256->512, KW=7) + bn6 + elu + avgpool
  k_pack_a<<<(512 * 1792) / 256, 256, 0, stream>>>(sep3_w, Abuf, 512, 1792, 512, 1792);
  k_pack_im2col<<<(2048 * 1792) / 256, 256, 0, stream>>>(h5, Btbuf, 256, 16, 7, 3);
  gemm(Abuf, Btbuf, Cbuf, 512, 2048, 1792);
  zstats();
  k_stats_cn_f32<<<512, 256, 0, stream>>>(Cbuf, 512, 2048, 16, subj, ssum, ssq);
  coeffs(bn6_g, bn6_b, 512, 2048.f);
  k_bn_apply_avgpool<<<(128 * 512 * 8) / 256, 256, 0, stream>>>(Cbuf, 512, 16, binv, bshift, subj, h6);

  // ---- attention (1x1, 512->512) + sigmoid gate
  k_pack_a<<<(512 * 512) / 256, 256, 0, stream>>>(attn_w, Abuf, 512, 512, 512, 512);
  k_pack_im2col<<<(1024 * 512) / 256, 256, 0, stream>>>(h6, Btbuf, 512, 8, 1, 0);
  gemm(Abuf, Btbuf, Cbuf, 512, 1024, 512);
  k_attn_apply<<<(128 * 512 * 8) / 256, 256, 0, stream>>>(Cbuf, h6, hatt);

  // ---- fc1 (4112 -> 1024, K padded to 4128) + elu
  k_pack_a<<<(1024 * 4128) / 256, 256, 0, stream>>>(fc1_w, Abuf, 1024, 4112, 1024, 4128);
  k_build_z0<<<(128 * 4128) / 256, 256, 0, stream>>>(hatt, embed, subj, Btbuf);
  gemm(Abuf, Btbuf, Cbuf, 1024, 128, 4128);
  k_bias_elu_t<<<(128 * 1024) / 256, 256, 0, stream>>>(Cbuf, fc1_b, Btbuf, 1024);

  // ---- fc2 (1024 -> 512) + elu
  k_pack_a<<<(512 * 1024) / 256, 256, 0, stream>>>(fc2_w, Abuf, 512, 1024, 512, 1024);
  gemm(Abuf, Btbuf, Cbuf, 512, 128, 1024);
  k_bias_elu_t<<<(128 * 512) / 256, 256, 0, stream>>>(Cbuf, fc2_b, Btbuf, 512);

  // ---- fc3 (512 -> 1854, M padded to 1856)
  k_pack_a<<<(1856 * 512) / 256, 256, 0, stream>>>(fc3_w, Abuf, 1854, 512, 1856, 512);
  gemm(Abuf, Btbuf, Cbuf, 1856, 128, 512);
  k_final_out<<<(128 * 1854 + 255) / 256, 256, 0, stream>>>(Cbuf, fc3_b, out);
}
